// multiHeadSelfAttention_31293131719527
// MI455X (gfx1250) — matrix operand order repaired and checked by reading, not compiled, not run on hardware
//
#include <hip/hip_runtime.h>

typedef _Float16 h8   __attribute__((ext_vector_type(8)));
typedef _Float16 h16  __attribute__((ext_vector_type(16)));
typedef float    f8   __attribute__((ext_vector_type(8)));
typedef unsigned int u32x4 __attribute__((ext_vector_type(4)));
typedef int      i32x4 __attribute__((ext_vector_type(4)));
typedef int      i32x8 __attribute__((ext_vector_type(8)));

#define SEQ    2048
#define DMODEL 1024
#define NHEAD  16
#define HDIM   64
#define MROWS  4096   // 2 batches * 2048 rows

// ---------------------------------------------------------------------------
// WMMA helper: D(16x16,f32) = A(16x32,f16) * B(32x16,f16) + C
// ---------------------------------------------------------------------------
__device__ __forceinline__ f8 wmma16(h16 a, h16 b, f8 c) {
  return __builtin_amdgcn_wmma_f32_16x16x32_f16(
      /*neg_a=*/false, a, /*neg_b=*/false, b,
      /*c_mod=*/(short)0, c, /*reuse_a=*/false, /*reuse_b=*/false);
}

// A-fragment (16x32, f16) from a row-major [rows x ld] source at tile origin.
// Layout: lanes 0-15 hold row m=lane, K=0..15; lanes 16-31 hold
// K=16..31.
__device__ __forceinline__ h16 load_a_frag(const _Float16* base, int ld, int lane) {
  const int m  = lane & 15;
  const int hf = lane >> 4;
  const _Float16* p = base + m * ld + hf * 16;
  h8 lo = *(const h8*)(p);
  h8 hi = *(const h8*)(p + 8);
  h16 r;
#pragma unroll
  for (int i = 0; i < 8; ++i) { r[i] = lo[i]; r[i + 8] = hi[i]; }
  return r;
}

// B-fragment (32x16, f16) for an NT operand: B[k][n] = src[n][k] with K
// contiguous. Lane n = lane&15 holds 16 contiguous K starting at (lane>>4)*16.
__device__ __forceinline__ h16 load_b_frag_nt(const _Float16* base, int ld, int lane) {
  const int n  = lane & 15;
  const int kh = lane >> 4;
  return *(const h16*)(base + n * ld + kh * 16);
}

// ---------------------------------------------------------------------------
// Tensor Data Mover: async 2D tile load (global f16, row-major) -> LDS.
// Builds the D# descriptor per CDNA5 ISA and issues
// tensor_load_to_lds. Tracked by TENSORcnt.
// ---------------------------------------------------------------------------
__device__ __forceinline__ void tdm_load_2d(const _Float16* gsrc, void* lds_dst,
                                            int tile_x, int tile_y, int stride_elems) {
  const unsigned long long ga = (unsigned long long)(uintptr_t)gsrc;
  const unsigned lds_off = (unsigned)(uintptr_t)lds_dst;  // LDS aperture: low 32 bits = offset

  u32x4 g0;
  g0[0] = 1u;                                        // count=1, user descriptor
  g0[1] = lds_off;                                   // lds_addr (bytes)
  g0[2] = (unsigned)(ga & 0xFFFFFFFFu);              // global_addr[31:0]
  g0[3] = (unsigned)((ga >> 32) & 0x01FFFFFFu)       // global_addr[56:32]
        | (2u << 30);                                // type = 2 ("image")

  const unsigned td0 = 1u << 20, td1 = 1u << 20;     // tensor dims (no OOB clipping)
  i32x8 g1;
  g1[0] = (int)(1u << 16);                           // data_size = 1 -> 2 bytes
  g1[1] = (int)((td0 & 0xFFFFu) << 16);              // tensor_dim0[15:0]
  g1[2] = (int)(((td0 >> 16) & 0xFFFFu) | ((td1 & 0xFFFFu) << 16));
  g1[3] = (int)(((td1 >> 16) & 0xFFFFu) | (((unsigned)tile_x & 0xFFFFu) << 16)); // tile_dim0
  g1[4] = (int)((unsigned)tile_y & 0xFFFFu);         // tile_dim1 (tile_dim2 = 0)
  g1[5] = stride_elems;                              // tensor_dim0_stride[31:0]
  g1[6] = 0;                                         // stride[47:32] | dim1_stride lo
  g1[7] = 0;

  i32x4 gz = {0, 0, 0, 0};
#if defined(__clang_major__) && (__clang_major__ >= 23)
  i32x8 gz8 = {0, 0, 0, 0, 0, 0, 0, 0};
  __builtin_amdgcn_tensor_load_to_lds(g0, g1, gz, gz, gz8, 0);
#else
  __builtin_amdgcn_tensor_load_to_lds(g0, g1, gz, gz, 0);
#endif
}

// ---------------------------------------------------------------------------
// f32 -> f16 conversion
// ---------------------------------------------------------------------------
__global__ void __launch_bounds__(256)
cvt_f32_f16(const float* __restrict__ src, _Float16* __restrict__ dst, int n) {
  int i = blockIdx.x * 256 + threadIdx.x;
  if (i < n) dst[i] = (_Float16)src[i];
}

// ---------------------------------------------------------------------------
// NT GEMM with TDM double-buffered LDS staging.
// out[m][n] = scale * sum_k A[m][k] * W[n][k];  M=4096, N=1024, K=1024 fixed.
// Block = 128(M) x 64(N); 8 waves, each wave a 16x64 tile, 4 accumulators.
// Per k-step (32): wave0 TDM-loads the 64x32 W tile, wave1 the 128x32 A tile
// into the ping-pong LDS buffers; WMMAs consume the previous buffers.
// MODE 0: f16 row-major; MODE 1: f16 V-transposed [bh][HDIM][SEQ];
// MODE 2: f32 + bias (final output).
// ---------------------------------------------------------------------------
template <int MODE>
__global__ void __launch_bounds__(256)
gemm_nt_f16(const _Float16* __restrict__ A, const _Float16* __restrict__ W,
            void* __restrict__ out, const float* __restrict__ bias, float scale) {
  constexpr int K = DMODEL, N = DMODEL, NK = K / 32;
  const int lane  = threadIdx.x & 31;
  const int wave  = threadIdx.x >> 5;
  const int row0  = blockIdx.y * 128;
  const int col0  = blockIdx.x * 64;
  const int mbase = (lane >> 4) * 8;
  const int n     = lane & 15;

  __shared__ __align__(32) _Float16 bbuf[2][64 * 32];    // W tile:  64 rows x 32 k
  __shared__ __align__(32) _Float16 abuf[2][128 * 32];   // A tile: 128 rows x 32 k

  // prologue: kick off k-step 0 transfers
  if (wave == 0)      tdm_load_2d(W + (size_t)col0 * K, &bbuf[0][0], 32, 64, K);
  else if (wave == 1) tdm_load_2d(A + (size_t)row0 * K, &abuf[0][0], 32, 128, K);

  f8 acc[4];
#pragma unroll
  for (int c = 0; c < 4; ++c)
#pragma unroll
    for (int r = 0; r < 8; ++r) acc[c][r] = 0.0f;

  for (int i = 0; i < NK; ++i) {
    const int cur = i & 1, nxt = cur ^ 1;
    if (wave == 0) {
      if (i + 1 < NK) {
        tdm_load_2d(W + (size_t)col0 * K + (i + 1) * 32, &bbuf[nxt][0], 32, 64, K);
        __builtin_amdgcn_s_wait_tensorcnt(1);   // oldest (cur) transfer retired
      } else {
        __builtin_amdgcn_s_wait_tensorcnt(0);
      }
    } else if (wave == 1) {
      if (i + 1 < NK) {
        tdm_load_2d(A + (size_t)row0 * K + (i + 1) * 32, &abuf[nxt][0], 32, 128, K);
        __builtin_amdgcn_s_wait_tensorcnt(1);
      } else {
        __builtin_amdgcn_s_wait_tensorcnt(0);
      }
    }
    __syncthreads();                            // publish cur buffers to all waves

    h16 a = load_a_frag(&abuf[cur][wave * 16 * 32], 32, lane);
#pragma unroll
    for (int c = 0; c < 4; ++c) {
      h16 b = load_b_frag_nt(&bbuf[cur][c * 16 * 32], 32, lane);
      acc[c] = wmma16(a, b, acc[c]);
    }
    __syncthreads();                            // reads done before nxt is re-filled
  }

  const int rw = row0 + wave * 16;              // this wave's 16 output rows
  if (MODE == 2) {
    float* o = (float*)out;
#pragma unroll
    for (int c = 0; c < 4; ++c) {
      const int col = col0 + c * 16 + n;
      const float bv = bias[col];
#pragma unroll
      for (int r = 0; r < 8; ++r)
        o[(size_t)(rw + mbase + r) * N + col] = acc[c][r] * scale + bv;
    }
  } else if (MODE == 1) {
    _Float16* o = (_Float16*)out;
#pragma unroll
    for (int c = 0; c < 4; ++c) {
      const int col = col0 + c * 16 + n;
      const int hh  = col >> 6;
      const int d   = col & 63;
#pragma unroll
      for (int r = 0; r < 8; ++r) {
        const int row = rw + mbase + r;
        const int bb  = row >> 11;
        const int s   = row & (SEQ - 1);
        o[((size_t)((bb * NHEAD + hh) * HDIM) + d) * SEQ + s] =
            (_Float16)(acc[c][r] * scale);
      }
    }
  } else {
    _Float16* o = (_Float16*)out;
#pragma unroll
    for (int c = 0; c < 4; ++c) {
      const int col = col0 + c * 16 + n;
#pragma unroll
      for (int r = 0; r < 8; ++r)
        o[(size_t)(rw + mbase + r) * N + col] = (_Float16)(acc[c][r] * scale);
    }
  }
}

// ---------------------------------------------------------------------------
// Flash attention forward (causal). One wave = 16 query rows of one (b,h).
// Q/K row-major [4096][1024] (Q pre-scaled by 1/8), V transposed [bh][64][2048].
// ---------------------------------------------------------------------------
__global__ void __launch_bounds__(256)
attn_fwd(const _Float16* __restrict__ Qh, const _Float16* __restrict__ Kh,
         const _Float16* __restrict__ Vt, _Float16* __restrict__ Ctx) {
  const int bh   = blockIdx.x;          // 0..31
  const int b    = bh >> 4;
  const int h    = bh & 15;
  const int lane = threadIdx.x & 31;
  const int wave = threadIdx.x >> 5;
  const int q0   = blockIdx.y * 128 + wave * 16;
  const int mbase = (lane >> 4) * 8;
  const int n     = lane & 15;

  __shared__ __align__(32) _Float16 pbuf[8][16 * 32];   // per-wave P staging

  h16 qa[2];
#pragma unroll
  for (int kc = 0; kc < 2; ++kc)
    qa[kc] = load_a_frag(Qh + (size_t)(b * SEQ + q0) * DMODEL + h * HDIM + kc * 32,
                         DMODEL, lane);

  f8 acc[4];
#pragma unroll
  for (int c = 0; c < 4; ++c)
#pragma unroll
    for (int r = 0; r < 8; ++r) acc[c][r] = 0.0f;

  float mstate[8], lstate[8];
#pragma unroll
  for (int r = 0; r < 8; ++r) { mstate[r] = -1e30f; lstate[r] = 0.0f; }

  for (int j = 0; j < q0 + 16; j += 32) {   // key blocks of 32, causal bound
    f8 s0, s1;
#pragma unroll
    for (int r = 0; r < 8; ++r) { s0[r] = 0.0f; s1[r] = 0.0f; }
#pragma unroll
    for (int kc = 0; kc < 2; ++kc) {
      h16 bk0 = load_b_frag_nt(
          Kh + (size_t)(b * SEQ + j) * DMODEL + h * HDIM + kc * 32, DMODEL, lane);
      h16 bk1 = load_b_frag_nt(
          Kh + (size_t)(b * SEQ + j + 16) * DMODEL + h * HDIM + kc * 32, DMODEL, lane);
      s0 = wmma16(qa[kc], bk0, s0);
      s1 = wmma16(qa[kc], bk1, s1);
    }

#pragma unroll
    for (int r = 0; r < 8; ++r) {
      const int row  = q0 + mbase + r;
      const bool k0m = (j + n) > row;
      const bool k1m = (j + 16 + n) > row;
      float v0 = k0m ? -1e30f : s0[r];
      float v1 = k1m ? -1e30f : s1[r];
      float mx = fmaxf(v0, v1);
#pragma unroll
      for (int off = 8; off > 0; off >>= 1)
        mx = fmaxf(mx, __shfl_xor(mx, off, 16));
      const float mnew  = fmaxf(mstate[r], mx);
      const float alpha = __expf(mstate[r] - mnew);
      const float p0 = k0m ? 0.0f : __expf(v0 - mnew);
      const float p1 = k1m ? 0.0f : __expf(v1 - mnew);
      float rs = p0 + p1;
#pragma unroll
      for (int off = 8; off > 0; off >>= 1)
        rs += __shfl_xor(rs, off, 16);
      lstate[r] = lstate[r] * alpha + rs;
      mstate[r] = mnew;
#pragma unroll
      for (int c = 0; c < 4; ++c) acc[c][r] *= alpha;
      pbuf[wave][(mbase + r) * 32 + n]      = (_Float16)p0;
      pbuf[wave][(mbase + r) * 32 + n + 16] = (_Float16)p1;
    }
    asm volatile("s_wait_dscnt 0" ::: "memory");  // wave-private LDS RAW

    h16 pa = load_a_frag(&pbuf[wave][0], 32, lane);
#pragma unroll
    for (int c = 0; c < 4; ++c) {
      h16 bv = load_b_frag_nt(
          Vt + ((size_t)(bh * HDIM) + c * 16) * SEQ + j, SEQ, lane);
      acc[c] = wmma16(pa, bv, acc[c]);
    }
  }

  float inv[8];
#pragma unroll
  for (int r = 0; r < 8; ++r) inv[r] = 1.0f / lstate[r];
#pragma unroll
  for (int c = 0; c < 4; ++c)
#pragma unroll
    for (int r = 0; r < 8; ++r)
      Ctx[(size_t)(b * SEQ + q0 + mbase + r) * DMODEL + h * HDIM + c * 16 + n] =
          (_Float16)(acc[c][r] * inv[r]);
}

// ---------------------------------------------------------------------------
// Launcher
// ---------------------------------------------------------------------------
extern "C" void kernel_launch(void* const* d_in, const int* in_sizes, int n_in,
                              void* d_out, int out_size, void* d_ws, size_t ws_size,
                              hipStream_t stream) {
  const float* x  = (const float*)d_in[0];
  const float* Wq = (const float*)d_in[1];
  const float* Wk = (const float*)d_in[2];
  const float* Wv = (const float*)d_in[3];
  const float* Wo = (const float*)d_in[4];
  const float* bo = (const float*)d_in[5];

  const int XN = MROWS * DMODEL;       // 4,194,304
  const int WN = DMODEL * DMODEL;      // 1,048,576

  _Float16* ws   = (_Float16*)d_ws;
  _Float16* xh   = ws;                 // x        f16 [4096][1024]
  _Float16* qh   = xh + XN;            // Q*0.125  f16 [4096][1024]
  _Float16* kh   = qh + XN;            // K        f16 [4096][1024]
  _Float16* vt   = kh + XN;            // V^T      f16 [32][64][2048]
  _Float16* ctxh = vt + XN;            // ctx      f16 [4096][1024]
  _Float16* wqh  = ctxh + XN;
  _Float16* wkh  = wqh + WN;
  _Float16* wvh  = wkh + WN;
  _Float16* woh  = wvh + WN;

  cvt_f32_f16<<<(XN + 255) / 256, 256, 0, stream>>>(x,  xh,  XN);
  cvt_f32_f16<<<(WN + 255) / 256, 256, 0, stream>>>(Wq, wqh, WN);
  cvt_f32_f16<<<(WN + 255) / 256, 256, 0, stream>>>(Wk, wkh, WN);
  cvt_f32_f16<<<(WN + 255) / 256, 256, 0, stream>>>(Wv, wvh, WN);
  cvt_f32_f16<<<(WN + 255) / 256, 256, 0, stream>>>(Wo, woh, WN);

  dim3 gg(DMODEL / 64, MROWS / 128);
  gemm_nt_f16<0><<<gg, 256, 0, stream>>>(xh, wqh, qh, nullptr, 0.125f);
  gemm_nt_f16<0><<<gg, 256, 0, stream>>>(xh, wkh, kh, nullptr, 1.0f);
  gemm_nt_f16<1><<<gg, 256, 0, stream>>>(xh, wvh, vt, nullptr, 1.0f);

  attn_fwd<<<dim3(2 * NHEAD, SEQ / 128), 256, 0, stream>>>(qh, kh, vt, ctxh);

  gemm_nt_f16<2><<<gg, 256, 0, stream>>>(ctxh, woh, d_out, bo, 1.0f);
}
